// RelativeAttention_83708912599209
// MI455X (gfx1250) — matrix operand order repaired and checked by reading, compiled, not run on hardware
//
#include <hip/hip_runtime.h>
#include <hip/hip_bf16.h>

typedef __bf16 bf16_t;
typedef __attribute__((ext_vector_type(16))) __bf16 v16bf;
typedef __attribute__((ext_vector_type(8)))  __bf16 v8bf;
typedef __attribute__((ext_vector_type(8)))  float  v8f;

// Problem constants
#define B_WIN  2048
#define SEQ    49
#define CH     512
#define HEADS  16
#define DH     32
#define NP     64          // SEQ padded to 4 WMMA row tiles
#define C3     1536

// LDS layout (bytes) -- total 295,936 of the 320KB WGP LDS
#define OFF_Q   0          // [16][64][32] bf16 = 65536
#define OFF_K   65536      // [16][64][32] bf16 = 65536
#define OFF_VT  131072     // [16][32][64] bf16 = 65536 (V transposed)
#define OFF_XA  196608     // [64][520]    bf16 = 66560 (X, then attn_out)
#define OFF_P   263168     // [16 waves][16][64] bf16 = 32768
#define XPITCH  520        // breaks 16-way bank conflict of 1024B row stride
#define LDS_BYTES 295936

// Workspace layout (bytes)
#define WS_WQKV 0          // 1536*512 bf16 = 1,572,864
#define WS_WPRJ 1572864    // 512*512 bf16  =   524,288
#define WS_BIAS 2097152    // 16*64*64 f32  =   262,144

__device__ __forceinline__ v8f zero8() {
  v8f z;
#pragma unroll
  for (int i = 0; i < 8; ++i) z[i] = 0.0f;
  return z;
}

__device__ __forceinline__ v16bf ld_pair(const bf16_t* p0, const bf16_t* p1) {
  v8bf a = *(const v8bf*)p0;
  v8bf b = *(const v8bf*)p1;
  v16bf r;
#pragma unroll
  for (int i = 0; i < 8; ++i) { r[i] = a[i]; r[i + 8] = b[i]; }
  return r;
}

// A fragment (16x32 tile of row-major [M][K]); lanes 0-15: K{0..7,16..23}, 16-31: K{8..15,24..31}
__device__ __forceinline__ v16bf fragA(const bf16_t* base, int pitch, int lane,
                                       int rowTile, int kBase) {
  const int row = rowTile + (lane & 15);
  const int klo = (lane < 16) ? 0 : 8;
  const bf16_t* p = base + row * pitch + kBase + klo;
  return ld_pair(p, p + 16);
}

// B fragment (32x16 tile), B^T stored row-major [N][K]; lanes 0-15: K{0..7,16..23}, 16-31: K{8..15,24..31}
__device__ __forceinline__ v16bf fragB(const bf16_t* base, int pitch, int lane,
                                       int colTile, int kBase) {
  const int col = colTile + (lane & 15);
  const int kh  = (lane < 16) ? 0 : 8;
  const bf16_t* p = base + col * pitch + kBase + kh;
  return ld_pair(p, p + 16);
}

__device__ __forceinline__ v8f wmma_bf16(v16bf a, v16bf b, v8f c) {
  return __builtin_amdgcn_wmma_f32_16x16x32_bf16(false, a, false, b, (short)0, c,
                                                 false, false);
}

// Wave-local LDS fence: DS ops from one wave execute in order at the LDS
// (ISA §7.1), so a same-wave cross-lane store->load only needs dscnt==0 plus
// a compiler reordering barrier -- no workgroup barrier.
__device__ __forceinline__ void lds_wave_fence() {
  asm volatile("s_wait_dscnt 0x0" ::: "memory");
}

// ---------------------------------------------------------------------------
// Prep: fp32 weights -> bf16; gather + pad relative-position bias to [16][64][64]
// (cols >= 49 get -1e30 so softmax over the padded 64 keys is exact)
// ---------------------------------------------------------------------------
__global__ void prep_kernel(const float* __restrict__ qkv_w,
                            const float* __restrict__ proj_w,
                            const float* __restrict__ bias_table,
                            const int*   __restrict__ rel_index,
                            bf16_t* __restrict__ wqkv,
                            bf16_t* __restrict__ wprj,
                            float*  __restrict__ biasPad) {
  const int idx = blockIdx.x * 256 + threadIdx.x;
  if (idx < C3 * CH)  wqkv[idx] = (bf16_t)qkv_w[idx];
  if (idx < CH * CH)  wprj[idx] = (bf16_t)proj_w[idx];
  if (idx < HEADS * NP * NP) {
    const int h   = idx >> 12;
    const int row = (idx >> 6) & 63;
    const int col = idx & 63;
    float bv;
    if (col >= SEQ)      bv = -1.0e30f;
    else if (row >= SEQ) bv = 0.0f;
    else                 bv = bias_table[rel_index[row * SEQ + col] * HEADS + h];
    biasPad[idx] = bv;
  }
}

// ---------------------------------------------------------------------------
// Fully fused window attention: one workgroup (16 waves) per window.
// ---------------------------------------------------------------------------
__global__ __launch_bounds__(512, 1)
void fused_attn_kernel(const float* __restrict__ x,
                       const float* __restrict__ qkv_b,
                       const float* __restrict__ proj_b,
                       const bf16_t* __restrict__ wqkv,
                       const bf16_t* __restrict__ wprj,
                       const float* __restrict__ biasPad,
                       float* __restrict__ out) {
  extern __shared__ char smem[];
  bf16_t* sQ  = (bf16_t*)(smem + OFF_Q);
  bf16_t* sK  = (bf16_t*)(smem + OFF_K);
  bf16_t* sVT = (bf16_t*)(smem + OFF_VT);
  bf16_t* sXA = (bf16_t*)(smem + OFF_XA);   // X in phase A; attn_out afterwards
  bf16_t* sP  = (bf16_t*)(smem + OFF_P);

  const int blk   = blockIdx.x;              // window id
  const int tid   = threadIdx.x;
  const int lane  = tid & 31;
  const int w     = tid >> 5;                // wave 0..15 (== head in phase B/C)
  const int colA  = lane & 15;
  const int rhalf = (lane >> 4) << 3;        // D-layout: lanes 16-31 hold rows +8

  // ---- stage X (fp32 -> bf16, zero row padding) ----
  for (int idx = tid; idx < NP * CH; idx += 512) {
    const int row = idx >> 9, col = idx & 511;
    const float v = (row < SEQ) ? x[((size_t)blk * SEQ + row) * CH + col] : 0.0f;
    sXA[row * XPITCH + col] = (bf16_t)v;
  }
  __syncthreads();

  // ---- Phase A: QKV = X @ Wqkv^T + b (wave w handles 6 of 96 column tiles) ----
  for (int c6 = 0; c6 < 6; ++c6) {
    const int jt   = (w * 6 + c6) * 16;
    const int jcol = jt + colA;
    const float bj = qkv_b[jcol];
    v8f acc[4];
#pragma unroll
    for (int rt = 0; rt < 4; ++rt) acc[rt] = zero8();
    for (int ks = 0; ks < CH / 32; ++ks) {
      const v16bf Bv = fragB(wqkv, CH, lane, jt, ks * 32);
      if (ks + 1 < CH / 32)  // pull next L2-resident weight line ahead
        __builtin_prefetch(wqkv + jcol * CH + (ks + 1) * 32, 0, 3);
#pragma unroll
      for (int rt = 0; rt < 4; ++rt) {
        const v16bf Av = fragA(sXA, XPITCH, lane, rt * 16, ks * 32);
        acc[rt] = wmma_bf16(Av, Bv, acc[rt]);
      }
    }
    const int   s   = jcol >> 9;            // 0=q 1=k 2=v
    const int   hh  = (jcol >> 5) & 15;
    const int   d   = jcol & 31;
    const float scl = (s == 0) ? 0.25f : 1.0f;  // H^-0.5 folded into Q
#pragma unroll
    for (int rt = 0; rt < 4; ++rt)
#pragma unroll
      for (int r = 0; r < 8; ++r) {
        const int row = rt * 16 + r + rhalf;
        const float v = (acc[rt][r] + bj) * scl;
        if (s == 0)      sQ [(hh * NP + row) * DH + d]  = (bf16_t)v;
        else if (s == 1) sK [(hh * NP + row) * DH + d]  = (bf16_t)v;
        else             sVT[(hh * DH + d) * NP + row]  = (bf16_t)v;  // transposed
      }
  }
  __syncthreads();

  // ---- Phase B+C: attention, wave w == head w (no cross-wave sync needed) ----
  {
    const int h = w;
    const bf16_t* q  = sQ  + h * (NP * DH);
    const bf16_t* kk = sK  + h * (NP * DH);
    const bf16_t* vt = sVT + h * (DH * NP);
    bf16_t* Pw = sP + w * (16 * NP);
    const float* bh = biasPad + h * (NP * NP);

    for (int rt = 0; rt < 4; ++rt) {
      // S row-block = Q_tile @ K^T (+bias): 4 col tiles, K-dim = dh = 32 (1 step)
      v8f sAcc[4];
      const v16bf Aq = fragA(q, DH, lane, rt * 16, 0);
#pragma unroll
      for (int ct = 0; ct < 4; ++ct)
        sAcc[ct] = wmma_bf16(Aq, fragB(kk, DH, lane, ct * 16, 0), zero8());
#pragma unroll
      for (int ct = 0; ct < 4; ++ct) {
        const float* bp = bh + (rt * 16 + rhalf) * NP + ct * 16 + colA;
#pragma unroll
        for (int r = 0; r < 8; ++r) sAcc[ct][r] += bp[r * NP];
      }
      // softmax over the 64 (padded) keys; row lives in one 16-lane half
#pragma unroll
      for (int r = 0; r < 8; ++r) {
        float m = fmaxf(fmaxf(sAcc[0][r], sAcc[1][r]), fmaxf(sAcc[2][r], sAcc[3][r]));
#pragma unroll
        for (int d = 1; d < 16; d <<= 1) m = fmaxf(m, __shfl_xor(m, d, 32));
        float e[4]; float sum = 0.0f;
#pragma unroll
        for (int ct = 0; ct < 4; ++ct) { e[ct] = __expf(sAcc[ct][r] - m); sum += e[ct]; }
#pragma unroll
        for (int d = 1; d < 16; d <<= 1) sum += __shfl_xor(sum, d, 32);
        const float inv = 1.0f / sum;
#pragma unroll
        for (int ct = 0; ct < 4; ++ct)
          Pw[(r + rhalf) * NP + ct * 16 + colA] = (bf16_t)(e[ct] * inv);
      }
      // Pw is wave-private: wave-local DS ordering is enough (no __syncthreads)
      lds_wave_fence();
      // O row-block = P @ V : N-dim = dh (2 tiles), K-dim = 64 (2 steps)
#pragma unroll
      for (int nt = 0; nt < 2; ++nt) {
        v8f o = zero8();
#pragma unroll
        for (int ks = 0; ks < 2; ++ks) {
          const v16bf Ap = fragA(Pw, NP, lane, 0, ks * 32);
          const v16bf Bv = fragB(vt, NP, lane, nt * 16, ks * 32);
          o = wmma_bf16(Ap, Bv, o);
        }
#pragma unroll
        for (int r = 0; r < 8; ++r) {
          const int row = rt * 16 + r + rhalf;
          sXA[row * XPITCH + h * DH + nt * 16 + colA] = (bf16_t)o[r];
        }
      }
    }
  }
  __syncthreads();

  // ---- Phase D: out = attn_out @ Wproj^T + b (wave w handles 2 of 32 col tiles) ----
#pragma unroll
  for (int c2 = 0; c2 < 2; ++c2) {
    const int ct  = (w * 2 + c2) * 16;
    const int col = ct + colA;
    const float pb = proj_b[col];
    v8f acc[4];
#pragma unroll
    for (int rt = 0; rt < 4; ++rt) acc[rt] = zero8();
    for (int ks = 0; ks < CH / 32; ++ks) {
      const v16bf Bv = fragB(wprj, CH, lane, ct, ks * 32);
      if (ks + 1 < CH / 32)
        __builtin_prefetch(wprj + col * CH + (ks + 1) * 32, 0, 3);
#pragma unroll
      for (int rt = 0; rt < 4; ++rt) {
        const v16bf Av = fragA(sXA, XPITCH, lane, rt * 16, ks * 32);
        acc[rt] = wmma_bf16(Av, Bv, acc[rt]);
      }
    }
#pragma unroll
    for (int rt = 0; rt < 4; ++rt)
#pragma unroll
      for (int r = 0; r < 8; ++r) {
        const int row = rt * 16 + r + rhalf;
        if (row < SEQ)
          out[((size_t)blk * SEQ + row) * CH + col] = acc[rt][r] + pb;
      }
  }
}

extern "C" void kernel_launch(void* const* d_in, const int* in_sizes, int n_in,
                              void* d_out, int out_size, void* d_ws, size_t ws_size,
                              hipStream_t stream) {
  (void)in_sizes; (void)n_in; (void)out_size; (void)ws_size;
  const float* x          = (const float*)d_in[0];
  const float* qkv_w      = (const float*)d_in[1];
  const float* qkv_b      = (const float*)d_in[2];
  const float* proj_w     = (const float*)d_in[3];
  const float* proj_b     = (const float*)d_in[4];
  const float* bias_table = (const float*)d_in[5];
  const int*   rel_index  = (const int*)d_in[6];
  float* out = (float*)d_out;

  char* ws = (char*)d_ws;
  bf16_t* wqkv   = (bf16_t*)(ws + WS_WQKV);
  bf16_t* wprj   = (bf16_t*)(ws + WS_WPRJ);
  float*  biasPd = (float*)(ws + WS_BIAS);

  prep_kernel<<<(C3 * CH + 255) / 256, 256, 0, stream>>>(
      qkv_w, proj_w, bias_table, rel_index, wqkv, wprj, biasPd);

  fused_attn_kernel<<<B_WIN, 512, LDS_BYTES, stream>>>(
      x, qkv_b, proj_b, wqkv, wprj, biasPd, out);
}
